// TwoStageAttentionLayer2_72249939853777
// MI455X (gfx1250) — compile-verified, matrix operand order checked by reading, not run on hardware
//
#include <hip/hip_runtime.h>
#include <hip/hip_bf16.h>
#include <math.h>

// ---------------------------------------------------------------------------
// TwoStageAttentionLayer (dead FourierBlock) == per-row fused pipeline:
//   y1 = LN(x + bo_time)        ; y2 = LN(y1 + MLPa(y1))
//   y3 = LN(y2 + bo_recv)       ; y4 = LN(y3 + MLPb(y3))  -> out (same layout)
// bf16 WMMA (v_wmma_f32_16x16x32_bf16) for the 128->512->128 GEMMs,
// f32 for LN / residuals; branch-free erf-GELU (A&S 7.1.26, |err|~1.5e-7).
// ---------------------------------------------------------------------------

typedef __attribute__((ext_vector_type(16))) __bf16 v16bf;
typedef __attribute__((ext_vector_type(2)))  __bf16 bf2;
typedef __attribute__((ext_vector_type(8)))  float  v8f;
typedef __attribute__((ext_vector_type(4)))  float  v4f;

#define DMODEL 128
#define DFF    512
#define ROWS   64          // rows per block
#define YSTR   132         // f32 LDS row stride (pad vs 128)
#define HSTR   520         // bf16 LDS row stride (pad vs 512)
#define ASTR   136         // bf16 LDS row stride (pad vs 128)
#define NMT    4           // ROWS/16 row-tiles

// K index inside a 32-K slab for (lane, elem i) of the 16-bit WMMA A/B operand
// (ISA 7.12.2, 16-bit A 16x32 layout; B mirrors it with lane = N column).
__device__ __forceinline__ int kmap(int lane, int i) {
  int v = i >> 1;
  return ((v >> 2) << 4) + (((lane >> 4) & 1) << 3) + ((v & 3) << 1) + (i & 1);
}

// Branch-free exact-erf GELU: 0.5*x*(1+erf(x/sqrt2)), A&S 7.1.26.
__device__ __forceinline__ float gelu_erf(float x) {
  float u  = x * 0.70710678118654752f;
  float au = __builtin_fabsf(u);
  float t  = __builtin_amdgcn_rcpf(__builtin_fmaf(0.3275911f, au, 1.0f));
  float p  = __builtin_fmaf(t, 1.061405429f, -1.453152027f);
  p = __builtin_fmaf(t, p, 1.421413741f);
  p = __builtin_fmaf(t, p, -0.284496736f);
  p = __builtin_fmaf(t, p, 0.254829592f);
  p = p * t;
  float e  = __builtin_amdgcn_exp2f(-au * au * 1.4426950408889634f);
  float er = __builtin_fmaf(-p, e, 1.0f);        // erf(|u|)
  er = __builtin_copysignf(er, u);
  return 0.5f * x * (1.0f + er);
}

// ---------------------------------------------------------------------------
// Weight prep: f32 row-major (K x N) -> bf16, WMMA-B swizzled:
// dst[((nTile*(K/32)+kSlab)*32 + lane)*16 + i] = W[k][n]
// ---------------------------------------------------------------------------
__global__ __launch_bounds__(256) void wprep_kernel(
    const float* __restrict__ W1a, const float* __restrict__ W2a,
    const float* __restrict__ W1b, const float* __restrict__ W2b,
    __bf16* __restrict__ dst)
{
  int idx = blockIdx.x * 256 + threadIdx.x;       // 0 .. 262143
  int mat = idx >> 16;
  int r   = idx & 0xFFFF;
  const float* W; int K, N;
  if      (mat == 0) { W = W1a; K = 128; N = 512; }
  else if (mat == 1) { W = W2a; K = 512; N = 128; }
  else if (mat == 2) { W = W1b; K = 128; N = 512; }
  else               { W = W2b; K = 512; N = 128; }
  int i     = r & 15;
  int lane  = (r >> 4) & 31;
  int slab  = r >> 9;                 // nTile*(K/32) + kSlab
  int kSlabs = K >> 5;
  int nTile = slab / kSlabs;
  int kSlab = slab - nTile * kSlabs;
  int k = kSlab * 32 + kmap(lane, i);
  int n = nTile * 16 + (lane & 15);
  dst[(size_t)mat * 65536 + r] = (__bf16)W[k * N + n];
}

// ---------------------------------------------------------------------------
struct TsaArgs {
  const float* x;
  const float* bo_time; const float* bo_recv;
  const float *g1, *be1, *g2, *be2, *g3, *be3, *g4, *be4;
  const float *b1a, *b2a, *b1b, *b2b;
  const __bf16 *w1a, *w2a, *w1b, *w2b;   // swizzled bf16 weights in d_ws
  float* out;
};

__global__ __launch_bounds__(256, 1) void tsa_kernel(TsaArgs A_)
{
  __shared__ float  yb [ROWS * YSTR];   // f32 activations (residual base)
  __shared__ __bf16 ybh[ROWS * ASTR];   // bf16 copy of activations (GEMM1 A)
  __shared__ float  zb [ROWS * YSTR];   // MLP output before residual+LN
  __shared__ __bf16 hb [ROWS * HSTR];   // GELU hidden, bf16 (GEMM2 A)

  const int tid  = threadIdx.x;
  const int lane = tid & 31;
  const int w    = tid >> 5;                    // wave id 0..7
  const long rowBase = (long)blockIdx.x * ROWS;

  // ---- load x tile (64 x 128 f32) into yb, coalesced b128 ----
  {
    const float* src = A_.x + rowBase * DMODEL;
    #pragma unroll
    for (int j = 0; j < 8; j++) {
      int fi   = tid + 256 * j;                 // float4 index in tile
      int row  = fi >> 5;                       // 32 float4 per row
      int col4 = (fi & 31) << 2;
      v4f v = *(const v4f*)(src + row * DMODEL + col4);
      yb[row * YSTR + col4 + 0] = v.x;
      yb[row * YSTR + col4 + 1] = v.y;
      yb[row * YSTR + col4 + 2] = v.z;
      yb[row * YSTR + col4 + 3] = v.w;
    }
  }
  __syncthreads();

  const int lrow = tid >> 2;                    // LN: 4 threads per row
  const int lq   = tid & 3;

  auto reduce4 = [&](float& s, float& s2) {
    s  += __shfl_xor(s,  1, 32);  s  += __shfl_xor(s,  2, 32);
    s2 += __shfl_xor(s2, 1, 32);  s2 += __shfl_xor(s2, 2, 32);
  };
  // write one normalized value pair to f32 + bf16 activation buffers
  auto store_act2 = [&](int c, float o0, float o1) {
    yb[lrow * YSTR + c]     = o0;
    yb[lrow * YSTR + c + 1] = o1;
    bf2 pp; pp.x = (__bf16)o0; pp.y = (__bf16)o1;
    *(bf2*)&ybh[lrow * ASTR + c] = pp;          // packed b32 store
  };

  // ---- dual-MLP GEMM via WMMA: ybh -> hb -> zb ----
  auto mlp = [&](const __bf16* w1s, const float* b1,
                 const __bf16* w2s, const float* b2) {
    const v16bf* w1v = (const v16bf*)w1s;
    const v16bf* w2v = (const v16bf*)w2s;
    __builtin_prefetch(w2s, 0, 1);              // warm W2 path in GL2

    // GEMM1: (64x128 bf16 ybh) x (128x512 W1) + b1 -> GELU -> hb (bf16)
    for (int m = 0; m < NMT; m++) {
      v16bf Aa[4];
      int arow = m * 16 + (lane & 15);
      #pragma unroll
      for (int ks = 0; ks < 4; ks++) {
        v16bf a;
        #pragma unroll
        for (int i = 0; i < 16; i += 2) {
          int k = ks * 32 + kmap(lane, i);
          bf2 pp = *(const bf2*)&ybh[arow * ASTR + k];
          a[i] = pp.x; a[i + 1] = pp.y;
        }
        Aa[ks] = a;
      }
      #pragma unroll
      for (int jj = 0; jj < 4; jj++) {
        int n = w + 8 * jj;                     // N-tile 0..31
        v8f c = {};
        #pragma unroll
        for (int ks = 0; ks < 4; ks++) {
          v16bf b = w1v[(n * 4 + ks) * 32 + lane];
          c = __builtin_amdgcn_wmma_f32_16x16x32_bf16(
                false, Aa[ks], false, b, (short)0, c, false, false);
        }
        float bias = b1[n * 16 + (lane & 15)];
        int mrow = m * 16 + ((lane >> 4) << 3);
        int ncol = n * 16 + (lane & 15);
        #pragma unroll
        for (int v = 0; v < 8; v++) {
          hb[(mrow + v) * HSTR + ncol] = (__bf16)gelu_erf(c[v] + bias);
        }
      }
    }
    __syncthreads();

    // GEMM2: (64x512 bf16 hb) x (512x128 W2) + b2 -> zb (f32)
    for (int m = 0; m < NMT; m++) {
      int n = w;                                // N-tile 0..7, one per wave
      int arow = m * 16 + (lane & 15);
      v8f c = {};
      #pragma unroll 4
      for (int ks = 0; ks < 16; ks++) {
        v16bf a;
        #pragma unroll
        for (int i = 0; i < 16; i += 2) {
          int k = ks * 32 + kmap(lane, i);
          bf2 pp = *(const bf2*)&hb[arow * HSTR + k];
          a[i] = pp.x; a[i + 1] = pp.y;
        }
        v16bf b = w2v[(n * 16 + ks) * 32 + lane];
        c = __builtin_amdgcn_wmma_f32_16x16x32_bf16(
              false, a, false, b, (short)0, c, false, false);
      }
      float bias = b2[n * 16 + (lane & 15)];
      int mrow = m * 16 + ((lane >> 4) << 3);
      int ncol = n * 16 + (lane & 15);
      #pragma unroll
      for (int v = 0; v < 8; v++) {
        zb[(mrow + v) * YSTR + ncol] = c[v] + bias;
      }
    }
    __syncthreads();
  };

  // ---- pass 1: y1 = LN(x + bo_time; g1,be1) -> yb/ybh ----
  {
    float vals[32]; float s = 0.f, s2 = 0.f;
    #pragma unroll
    for (int j = 0; j < 32; j++) {
      int c = lq * 32 + j;
      float v = yb[lrow * YSTR + c] + A_.bo_time[c];
      vals[j] = v; s += v; s2 += v * v;
    }
    reduce4(s, s2);
    float mean = s * (1.f / 128.f);
    float rinv = rsqrtf(s2 * (1.f / 128.f) - mean * mean + 1e-5f);
    #pragma unroll
    for (int j = 0; j < 32; j += 2) {
      int c = lq * 32 + j;
      float o0 = (vals[j]     - mean) * rinv * A_.g1[c]     + A_.be1[c];
      float o1 = (vals[j + 1] - mean) * rinv * A_.g1[c + 1] + A_.be1[c + 1];
      store_act2(c, o0, o1);
    }
  }
  __syncthreads();

  mlp(A_.w1a, A_.b1a, A_.w2a, A_.b2a);

  // ---- pass 2 (fused): y2 = LN(y1+z; g2,be2); y3 = LN(y2+bo_recv; g3,be3) ----
  {
    float vals[32]; float s = 0.f, s2 = 0.f;
    #pragma unroll
    for (int j = 0; j < 32; j++) {
      int c = lq * 32 + j;
      float v = yb[lrow * YSTR + c] + zb[lrow * YSTR + c];
      vals[j] = v; s += v; s2 += v * v;
    }
    reduce4(s, s2);
    float mean = s * (1.f / 128.f);
    float rinv = rsqrtf(s2 * (1.f / 128.f) - mean * mean + 1e-5f);
    float t = 0.f, t2 = 0.f;
    #pragma unroll
    for (int j = 0; j < 32; j++) {
      int c = lq * 32 + j;
      float y2 = (vals[j] - mean) * rinv * A_.g2[c] + A_.be2[c];
      float v3 = y2 + A_.bo_recv[c];
      vals[j] = v3; t += v3; t2 += v3 * v3;
    }
    reduce4(t, t2);
    float mean3 = t * (1.f / 128.f);
    float rinv3 = rsqrtf(t2 * (1.f / 128.f) - mean3 * mean3 + 1e-5f);
    #pragma unroll
    for (int j = 0; j < 32; j += 2) {
      int c = lq * 32 + j;
      float o0 = (vals[j]     - mean3) * rinv3 * A_.g3[c]     + A_.be3[c];
      float o1 = (vals[j + 1] - mean3) * rinv3 * A_.g3[c + 1] + A_.be3[c + 1];
      store_act2(c, o0, o1);
    }
  }
  __syncthreads();

  mlp(A_.w1b, A_.b1b, A_.w2b, A_.b2b);

  // ---- pass 3: y4 = LN(y3 + z; g4,be4) -> out (same b,d,l,dm layout) ----
  {
    float vals[32]; float s = 0.f, s2 = 0.f;
    #pragma unroll
    for (int j = 0; j < 32; j++) {
      int c = lq * 32 + j;
      float v = yb[lrow * YSTR + c] + zb[lrow * YSTR + c];
      vals[j] = v; s += v; s2 += v * v;
    }
    reduce4(s, s2);
    float mean = s * (1.f / 128.f);
    float rinv = rsqrtf(s2 * (1.f / 128.f) - mean * mean + 1e-5f);
    float* orow = A_.out + (rowBase + lrow) * (long)DMODEL + lq * 32;
    #pragma unroll
    for (int j = 0; j < 32; j++) {
      int c = lq * 32 + j;
      orow[j] = (vals[j] - mean) * rinv * A_.g4[c] + A_.be4[c];
    }
  }
}

// ---------------------------------------------------------------------------
extern "C" void kernel_launch(void* const* d_in, const int* in_sizes, int n_in,
                              void* d_out, int out_size, void* d_ws, size_t ws_size,
                              hipStream_t stream)
{
  (void)in_sizes; (void)n_in; (void)out_size; (void)ws_size;
  const float* x       = (const float*)d_in[0];
  // d_in[1] = router (dead), d_in[3] = bo_send (dead)
  const float* bo_time = (const float*)d_in[2];
  const float* bo_recv = (const float*)d_in[4];
  const float* g1  = (const float*)d_in[5];  const float* be1 = (const float*)d_in[6];
  const float* g2  = (const float*)d_in[7];  const float* be2 = (const float*)d_in[8];
  const float* g3  = (const float*)d_in[9];  const float* be3 = (const float*)d_in[10];
  const float* g4  = (const float*)d_in[11]; const float* be4 = (const float*)d_in[12];
  const float* W1a = (const float*)d_in[13]; const float* b1a = (const float*)d_in[14];
  const float* W2a = (const float*)d_in[15]; const float* b2a = (const float*)d_in[16];
  const float* W1b = (const float*)d_in[17]; const float* b1b = (const float*)d_in[18];
  const float* W2b = (const float*)d_in[19]; const float* b2b = (const float*)d_in[20];

  __bf16* wz = (__bf16*)d_ws;                  // 4 * 65536 bf16 = 512 KB
  wprep_kernel<<<1024, 256, 0, stream>>>(W1a, W2a, W1b, W2b, wz);

  TsaArgs a;
  a.x = x; a.bo_time = bo_time; a.bo_recv = bo_recv;
  a.g1 = g1; a.be1 = be1; a.g2 = g2; a.be2 = be2;
  a.g3 = g3; a.be3 = be3; a.g4 = g4; a.be4 = be4;
  a.b1a = b1a; a.b2a = b2a; a.b1b = b1b; a.b2b = b2b;
  a.w1a = wz;           a.w2a = wz + 65536;
  a.w1b = wz + 131072;  a.w2b = wz + 196608;
  a.out = (float*)d_out;

  // 131072 rows / 64 rows per block
  tsa_kernel<<<2048, 256, 0, stream>>>(a);
}
